// h_RGLSTM_15272903704603
// MI455X (gfx1250) — matrix operand order repaired and checked by reading, not compiled, not run on hardware
//
#include <hip/hip_runtime.h>
#include <stdint.h>

// Problem dims (fixed by reference): B=64, S=512, I=1024, H=1024
#define Bb 64
#define Ss 512
#define Ii 1024
#define Hh 1024

typedef __attribute__((ext_vector_type(16))) __bf16 v16bf;
typedef __attribute__((ext_vector_type(8)))  float  v8f;
typedef __attribute__((ext_vector_type(4)))  int    v4i;

#ifndef __has_builtin
#define __has_builtin(x) 0
#endif

#if __has_builtin(__builtin_amdgcn_global_load_async_to_lds_b128)
#define USE_ASYNC_LDS 1
#else
#define USE_ASYNC_LDS 0
#endif

// Builtin signature (from hipcc diagnostic): param0 is int4* in AS(1) (global),
// param1 is the LDS-side pointer in AS(3); both non-const.
typedef __attribute__((address_space(1))) v4i gv4i;
typedef __attribute__((address_space(3))) v4i lv4i;

// ---------- helpers ----------

__device__ __forceinline__ uint16_t f2bf(float f) {
  uint32_t u = __builtin_bit_cast(uint32_t, f);
  u += 0x7FFFu + ((u >> 16) & 1u);      // round-to-nearest-even
  return (uint16_t)(u >> 16);
}

struct Frag32B { uint4 lo, hi; };

// A fragment: 16x32 bf16, row-major source [rows][ld], tile at (row0, k0).
// ISA layout: lane m=lane&15, half=lane>>4; VGPR0-3 = K {half*8 .. +7},
// VGPR4-7 = K {16+half*8 .. +7}. Two contiguous 16B loads per lane.
// Works for both global and LDS source pointers (LDS -> ds_load_b128).
__device__ __forceinline__ v16bf load_a_frag(const uint16_t* M,
                                             int ld, int row0, int k0, int lane) {
  int m = lane & 15, half = lane >> 4;
  const uint16_t* p = M + (size_t)(row0 + m) * ld + k0 + half * 8;
  Frag32B r;
  r.lo = *(const uint4*)(p);
  r.hi = *(const uint4*)(p + 16);
  return __builtin_bit_cast(v16bf, r);
}

// B fragment: 32(K)x16(N) bf16 where B[k][n] = W[n][k], W row-major [N][ld].
// Lane n=lane&15 holds K {half*8 .. +7} then K {16+half*8 .. +7} (matches A).
__device__ __forceinline__ v16bf load_b_frag(const uint16_t* __restrict__ W,
                                             int ld, int n0, int k0, int lane) {
  int n = lane & 15, half = lane >> 4;
  const uint16_t* p = W + (size_t)(n0 + n) * ld + k0 + half * 8;
  Frag32B r;
  r.lo = *(const uint4*)(p);
  r.hi = *(const uint4*)(p + 16);
  return __builtin_bit_cast(v16bf, r);
}

__device__ __forceinline__ float sigmoidf(float x) {
  return 1.0f / (1.0f + __expf(-x));
}
__device__ __forceinline__ float softplusf(float x) {
  return x > 20.0f ? x : log1pf(__expf(x));
}

#define WMMA_BF16(A, Bm, C) \
  __builtin_amdgcn_wmma_f32_16x16x32_bf16(false, (A), false, (Bm), (short)0, (C), false, false)

// Cooperatively stage `nbytes` (multiple of 4096) from global into LDS.
// Uses gfx1250 async global->LDS b128 DMA when available; waits on ASYNCcnt
// then barriers so all waves see the staged tile.
__device__ __forceinline__ void stage_to_lds(const uint16_t* __restrict__ g,
                                             uint16_t* l, int nbytes) {
  int tid = threadIdx.x;
#if USE_ASYNC_LDS
  for (int off = tid * 16; off < nbytes; off += 256 * 16) {
    __builtin_amdgcn_global_load_async_to_lds_b128(
        (gv4i*)((const char*)g + off),
        (lv4i*)((char*)l + off), 0, 0);
  }
#if __has_builtin(__builtin_amdgcn_s_wait_asynccnt)
  __builtin_amdgcn_s_wait_asynccnt(0);
#else
  asm volatile("s_wait_asynccnt 0x0" ::: "memory");
#endif
#else
  for (int off = tid * 16; off < nbytes; off += 256 * 16) {
    *(uint4*)((char*)l + off) = *(const uint4*)((const char*)g + off);
  }
#endif
  __syncthreads();
}

// ---------- one-time conversion kernels ----------

__global__ void convert_bf16_kernel(const float* __restrict__ in,
                                    uint16_t* __restrict__ out, int n) {
  int i = blockIdx.x * blockDim.x + threadIdx.x;
  if (i < n) out[i] = f2bf(in[i]);
}

// W_gauss is (Ii=1024, 2H=2048); produce WgT (2048, 1024) bf16 so B-frags are contiguous.
__global__ void transpose_wgauss_kernel(const float* __restrict__ in,
                                        uint16_t* __restrict__ out) {
  int i = blockIdx.x * blockDim.x + threadIdx.x;   // over 2048*1024
  if (i >= 2 * Hh * Ii) return;
  int n = i / Ii, k = i % Ii;
  out[i] = f2bf(in[(size_t)k * (2 * Hh) + n]);
}

// ---------- big parallel GEMM: gpre = x @ W_ih^T + (b_ih + b_hh) ----------
// x2d: (B*S, I) bf16 ; W_ih: (4H, I) bf16 ; gpre: (B*S, 4H) fp32
__global__ void __launch_bounds__(256)
pregemm_kernel(const uint16_t* __restrict__ xbf,
               const uint16_t* __restrict__ Wih,
               const float* __restrict__ bih,
               const float* __restrict__ bhh,
               float* __restrict__ gpre) {
  int lane = threadIdx.x & 31;
  int wave = threadIdx.x >> 5;
  int n0 = blockIdx.x * 16;                 // 256 blocks over 4H
  int m0 = blockIdx.y * 128 + wave * 16;    // 256 blocks * 8 waves over B*S
  v8f acc = {};
  for (int k = 0; k < Ii; k += 32) {
    v16bf a = load_a_frag(xbf, Ii, m0, k, lane);
    v16bf b = load_b_frag(Wih, Ii, n0, k, lane);
    acc = WMMA_BF16(a, b, acc);
  }
  int col  = n0 + (lane & 15);
  int half = lane >> 4;
  float bias = bih[col] + bhh[col];
#pragma unroll
  for (int r = 0; r < 8; ++r) {
    int m = m0 + r + half * 8;
    gpre[(size_t)m * (4 * Hh) + col] = acc[r] + bias;
  }
}

// ---------- initial sampler (h0 = 0 -> stats = 0) ----------
__global__ void init_kernel(const float* __restrict__ eps,   // eps[0] slice
                            float* __restrict__ out_mus,
                            float* __restrict__ out_stds,
                            uint16_t* __restrict__ hbf,
                            float* __restrict__ cbuf) {
  int idx = blockIdx.x * blockDim.x + threadIdx.x;
  if (idx >= Bb * Hh) return;
  int m = idx / Hh, col = idx % Hh;
  const float mu0  = 1e-6f;                   // clip(0, 1e-6, 1e6)
  const float std0 = 0.69314718055994531f;    // softplus(0)
  float hn = mu0 + eps[idx] * std0;
  size_t o = (size_t)m * (Ss + 1) * Hh + col; // t index 0
  out_mus[o]  = mu0;
  out_stds[o] = std0;
  hbf[idx]  = f2bf(hn);
  cbuf[idx] = 0.0f;
}

// ---------- per-step fused gate GEMM + LSTM pointwise ----------
// 8 blocks x 8 waves. Wave owns one 16-col n-tile; computes ALL 4 batch
// m-tiles and ALL 4 gates for it: each B-fragment feeds 4 WMMAs, each k-step
// issues 16 independent WMMA chains. h (64x1024 bf16 = 128KB) is staged in
// LDS once per block via async global->LDS DMA; A-frags come from ds_load.
__global__ void __launch_bounds__(256)
gate_step_kernel(const uint16_t* __restrict__ hbf,   // (B,H) bf16 current h
                 const uint16_t* __restrict__ Whh,   // (4H,H) bf16
                 const float* __restrict__ gpre,     // (B*S,4H) fp32
                 float* __restrict__ cbuf,           // (B,H) fp32
                 uint16_t* __restrict__ hraw,        // (B,H) bf16 out
                 float* __restrict__ out_clast,
                 int t) {
  extern __shared__ __align__(16) uint16_t lds_h[];  // Bb*Hh bf16 = 128KB
  int lane = threadIdx.x & 31;
  int wave = threadIdx.x >> 5;
  int n0 = (blockIdx.x * 8 + wave) * 16;             // 64 n-tiles total

  stage_to_lds(hbf, lds_h, Bb * Hh * 2);

  v8f acc[4][4] = {};                                // [gate][m-tile]
  for (int k = 0; k < Hh; k += 32) {
    v16bf a[4];
#pragma unroll
    for (int mt = 0; mt < 4; ++mt)
      a[mt] = load_a_frag(lds_h, Hh, mt * 16, k, lane);
#pragma unroll
    for (int gt = 0; gt < 4; ++gt) {
      v16bf b = load_b_frag(Whh, Hh, n0 + gt * Hh, k, lane);
#pragma unroll
      for (int mt = 0; mt < 4; ++mt)
        acc[gt][mt] = WMMA_BF16(a[mt], b, acc[gt][mt]);
    }
  }

  int col  = n0 + (lane & 15);
  int half = lane >> 4;
#pragma unroll
  for (int mt = 0; mt < 4; ++mt) {
#pragma unroll
    for (int r = 0; r < 8; ++r) {
      int m = mt * 16 + r + half * 8;
      size_t grow = ((size_t)m * Ss + t) * (4 * Hh);
      float iv = sigmoidf(acc[0][mt][r] + gpre[grow + col]);
      float fv = sigmoidf(acc[1][mt][r] + gpre[grow + Hh + col]);
      float gv = tanhf  (acc[2][mt][r] + gpre[grow + 2 * Hh + col]);
      float ov = sigmoidf(acc[3][mt][r] + gpre[grow + 3 * Hh + col]);
      size_t ci = (size_t)m * Hh + col;
      float cn = fv * cbuf[ci] + iv * gv;
      cbuf[ci] = cn;
      hraw[ci] = f2bf(ov * tanhf(cn));
      if (t == Ss - 1) out_clast[ci] = cn;
    }
  }
}

// ---------- per-step fused sampler GEMM + reparameterization ----------
// Same structure: wave owns cols [n0,n0+16) of BOTH the mu half and the
// softplus half of stats, for all 4 m-tiles. h_raw staged in LDS per block.
__global__ void __launch_bounds__(256)
sampler_step_kernel(const uint16_t* __restrict__ hraw,  // (B,H) bf16
                    const uint16_t* __restrict__ WgT,   // (2H,H) bf16
                    const float* __restrict__ eps,      // (S+1,B,H)
                    float* __restrict__ out_hidden,     // (B,S,H)
                    float* __restrict__ out_mus,        // (B,S+1,H)
                    float* __restrict__ out_stds,       // (B,S+1,H)
                    float* __restrict__ out_hlast,      // (B,H)
                    uint16_t* __restrict__ hbf,         // next-step h bf16
                    int t) {
  extern __shared__ __align__(16) uint16_t lds_h[];
  int lane = threadIdx.x & 31;
  int wave = threadIdx.x >> 5;
  int n0 = (blockIdx.x * 8 + wave) * 16;

  stage_to_lds(hraw, lds_h, Bb * Hh * 2);

  v8f amu[4] = {}, asp[4] = {};
  for (int k = 0; k < Hh; k += 32) {
    v16bf a[4];
#pragma unroll
    for (int mt = 0; mt < 4; ++mt)
      a[mt] = load_a_frag(lds_h, Hh, mt * 16, k, lane);
    v16bf bmu = load_b_frag(WgT, Hh, n0, k, lane);
    v16bf bsp = load_b_frag(WgT, Hh, Hh + n0, k, lane);
#pragma unroll
    for (int mt = 0; mt < 4; ++mt) {
      amu[mt] = WMMA_BF16(a[mt], bmu, amu[mt]);
      asp[mt] = WMMA_BF16(a[mt], bsp, asp[mt]);
    }
  }

  int col  = n0 + (lane & 15);
  int half = lane >> 4;
#pragma unroll
  for (int mt = 0; mt < 4; ++mt) {
#pragma unroll
    for (int r = 0; r < 8; ++r) {
      int m = mt * 16 + r + half * 8;
      float mu = fminf(fmaxf(amu[mt][r], 1e-6f), 1e6f);
      float st = fmaxf(softplusf(asp[mt][r]), 1e-6f);
      float e  = eps[((size_t)(t + 1) * Bb + m) * Hh + col];
      float hn = mu + e * st;
      out_hidden[((size_t)m * Ss + t) * Hh + col] = hn;
      size_t o = ((size_t)m * (Ss + 1) + (t + 1)) * Hh + col;
      out_mus[o]  = mu;
      out_stds[o] = st;
      hbf[(size_t)m * Hh + col] = f2bf(hn);
      if (t == Ss - 1) out_hlast[(size_t)m * Hh + col] = hn;
    }
  }
}

// ---------- launcher ----------

extern "C" void kernel_launch(void* const* d_in, const int* in_sizes, int n_in,
                              void* d_out, int out_size, void* d_ws, size_t ws_size,
                              hipStream_t stream) {
  (void)in_sizes; (void)n_in; (void)out_size; (void)ws_size;

  const float* x   = (const float*)d_in[0];
  const float* Wih = (const float*)d_in[1];
  const float* Whh = (const float*)d_in[2];
  const float* bih = (const float*)d_in[3];
  const float* bhh = (const float*)d_in[4];
  const float* Wg  = (const float*)d_in[5];
  const float* eps = (const float*)d_in[6];

  float* out        = (float*)d_out;
  float* out_hidden = out;                                   // (B,S,H)
  float* out_mus    = out_hidden + (size_t)Bb * Ss * Hh;     // (B,S+1,H)
  float* out_stds   = out_mus + (size_t)Bb * (Ss + 1) * Hh;  // (B,S+1,H)
  float* out_hlast  = out_stds + (size_t)Bb * (Ss + 1) * Hh; // (B,H)
  float* out_clast  = out_hlast + (size_t)Bb * Hh;           // (B,H)

  char* ws = (char*)d_ws;
  float*    gpre  = (float*)ws;    ws += (size_t)Bb * Ss * 4 * Hh * 4;
  uint16_t* xbf   = (uint16_t*)ws; ws += (size_t)Bb * Ss * Ii * 2;
  uint16_t* Wihb  = (uint16_t*)ws; ws += (size_t)4 * Hh * Ii * 2;
  uint16_t* Whhb  = (uint16_t*)ws; ws += (size_t)4 * Hh * Hh * 2;
  uint16_t* WgT   = (uint16_t*)ws; ws += (size_t)2 * Hh * Ii * 2;
  uint16_t* hbf   = (uint16_t*)ws; ws += (size_t)Bb * Hh * 2;
  uint16_t* hraw  = (uint16_t*)ws; ws += (size_t)Bb * Hh * 2;
  float*    cbuf  = (float*)ws;    ws += (size_t)Bb * Hh * 4;

  // One-time conversions to bf16 (WMMA-friendly layouts).
  int nx = Bb * Ss * Ii;
  convert_bf16_kernel<<<(nx + 255) / 256, 256, 0, stream>>>(x, xbf, nx);
  int nw = 4 * Hh * Ii;
  convert_bf16_kernel<<<(nw + 255) / 256, 256, 0, stream>>>(Wih, Wihb, nw);
  convert_bf16_kernel<<<(nw + 255) / 256, 256, 0, stream>>>(Whh, Whhb, nw);
  int ng = 2 * Hh * Ii;
  transpose_wgauss_kernel<<<(ng + 255) / 256, 256, 0, stream>>>(Wg, WgT);

  // Batch-parallel input projection (275 GFLOP, fully parallel).
  pregemm_kernel<<<dim3(4 * Hh / 16, (Bb * Ss) / 128), 256, 0, stream>>>(
      xbf, Wihb, bih, bhh, gpre);

  // Initial sample from h0 = 0.
  init_kernel<<<(Bb * Hh + 255) / 256, 256, 0, stream>>>(eps, out_mus, out_stds,
                                                         hbf, cbuf);

  // Sequential recurrence: 2 fused WMMA kernels per step (graph-replayed).
  const size_t lds_bytes = (size_t)Bb * Hh * 2;  // 128 KB dynamic LDS
  for (int t = 0; t < Ss; ++t) {
    gate_step_kernel<<<8, 256, lds_bytes, stream>>>(hbf, Whhb, gpre, cbuf, hraw,
                                                    out_clast, t);
    sampler_step_kernel<<<8, 256, lds_bytes, stream>>>(hraw, WgT, eps, out_hidden,
                                                       out_mus, out_stds, out_hlast,
                                                       hbf, t);
  }
}
